// ProdLDA_77661598646345
// MI455X (gfx1250) — matrix operand order repaired and checked by reading, not compiled, not run on hardware
//
#include <hip/hip_runtime.h>
#include <hip/hip_bf16.h>
#include <math.h>

// ---------------------------------------------------------------------------
// ProdLDA forward for MI455X (gfx1250, wave32, WMMA).
// Heavy GEMMs: v_wmma_f32_16x16x32_bf16, bf16 A/B converted in-register,
// fp32 accumulate. B operands are consumed transposed (contiguous-K rows) so
// all fragment loads are unconditional 16B vector loads; out-of-range rows /
// cols are handled by address clamping (their garbage never gets stored).
// ---------------------------------------------------------------------------

typedef __attribute__((ext_vector_type(16))) __bf16 v16bf;
typedef __attribute__((ext_vector_type(8)))  float  v8f;
typedef float f32x4 __attribute__((ext_vector_type(4), aligned(4)));

#define EPI_NONE          0
#define EPI_BIAS_RELU     1
#define EPI_BIAS_RELU_RES 2

__device__ __forceinline__ int imin(int a, int b) { return a < b ? a : b; }

// A fragment (16-bit A 16x32, contiguous K order used here, same as B):
//   lanes 0-15 : row=lane,    elems 0..15 -> K=k0+{0..15}
//   lanes16-31 : row=lane-16, elems 0..15 -> K=k0+{16..31}
// (ksel = (lane>>4)*16 folded into the row pointer offset)
template <bool GUARD>
__device__ __forceinline__ v16bf load_a_frag(const float* __restrict__ arow,
                                             int k0, int ksel, int K) {
    v16bf af;
    if (!GUARD) {
        const f32x4 a0 = *(const f32x4*)(arow + k0 + ksel);
        const f32x4 a1 = *(const f32x4*)(arow + k0 + ksel + 4);
        const f32x4 a2 = *(const f32x4*)(arow + k0 + ksel + 8);
        const f32x4 a3 = *(const f32x4*)(arow + k0 + ksel + 12);
#pragma unroll
        for (int j = 0; j < 4; ++j) {
            af[j]      = (__bf16)a0[j];
            af[j + 4]  = (__bf16)a1[j];
            af[j + 8]  = (__bf16)a2[j];
            af[j + 12] = (__bf16)a3[j];
        }
    } else {
#pragma unroll
        for (int j = 0; j < 8; ++j) {
            const int ka = k0 + ksel + j;
            const int kb = ka + 8;
            const float va = arow[imin(ka, K - 1)];
            const float vb = arow[imin(kb, K - 1)];
            af[j]     = (__bf16)(ka < K ? va : 0.0f);
            af[j + 8] = (__bf16)(kb < K ? vb : 0.0f);
        }
    }
    return af;
}

// B fragment (32x16) from a TRANSPOSED B (BT row-major [Nc, K]):
//   lanes 0-15 : col=lane,    K=k0+{0..15};  lanes16-31 : col=lane-16, K=k0+{16..31}
template <bool GUARD>
__device__ __forceinline__ v16bf load_b_frag(const float* __restrict__ brow,
                                             int k0, int ksel, int K) {
    v16bf bf;
    if (!GUARD) {
        const f32x4 b0 = *(const f32x4*)(brow + k0 + ksel);
        const f32x4 b1 = *(const f32x4*)(brow + k0 + ksel + 4);
        const f32x4 b2 = *(const f32x4*)(brow + k0 + ksel + 8);
        const f32x4 b3 = *(const f32x4*)(brow + k0 + ksel + 12);
#pragma unroll
        for (int j = 0; j < 4; ++j) {
            bf[j]      = (__bf16)b0[j];
            bf[j + 4]  = (__bf16)b1[j];
            bf[j + 8]  = (__bf16)b2[j];
            bf[j + 12] = (__bf16)b3[j];
        }
    } else {
#pragma unroll
        for (int j = 0; j < 16; ++j) {
            const int kk = k0 + ksel + j;
            const float v = brow[imin(kk, K - 1)];
            bf[j] = (__bf16)(kk < K ? v : 0.0f);
        }
    }
    return bf;
}

#define WMMA_BF16(AF, BF, ACC) \
    __builtin_amdgcn_wmma_f32_16x16x32_bf16(false, (AF), false, (BF), (short)0, (ACC), false, false)

// One wave computes a 16x32 C block (two 16x16 WMMA tiles sharing the A frag).
//   C = A[M,K] * B[K,Nc] (+epilogue);  B passed TRANSPOSED: BT [Nc, K].
//   If STORE_T: C written transposed as CT [Nc, M] (contiguous per lane).
template <int EPI, bool STORE_T>
__global__ __launch_bounds__(32) void gemm_bf16_wmma(
    const float* __restrict__ A, const float* __restrict__ BT,
    float* __restrict__ C,
    const float* __restrict__ bias, const float* __restrict__ residual,
    int M, int K, int Nc, long strideA, long strideBT, long strideC)
{
    const int lane = threadIdx.x;
    const int row0 = blockIdx.x * 16;
    const int col0 = blockIdx.y * 32;
    const int b    = blockIdx.z;

    const int r  = imin(row0 + (lane & 15), M - 1);        // clamp: garbage rows unstored
    const int c0 = imin(col0 + (lane & 15), Nc - 1);       // clamp: garbage cols unstored
    const int c1 = imin(col0 + 16 + (lane & 15), Nc - 1);
    const int kselA = (lane >> 4) * 16;
    const int kselB = (lane >> 4) * 16;

    const float* __restrict__ arow  = A  + (long)b * strideA  + (long)r  * K;
    const float* __restrict__ brow0 = BT + (long)b * strideBT + (long)c0 * K;
    const float* __restrict__ brow1 = BT + (long)b * strideBT + (long)c1 * K;

    v8f acc0 = {}, acc1 = {};
    const int Kfull = K & ~31;
    for (int k0 = 0; k0 < Kfull; k0 += 32) {
        __builtin_prefetch(arow + k0 + 128, 0, 0);  // speculative stream-ahead on A
        const v16bf af  = load_a_frag<false>(arow,  k0, kselA, K);
        const v16bf bf0 = load_b_frag<false>(brow0, k0, kselB, K);
        const v16bf bf1 = load_b_frag<false>(brow1, k0, kselB, K);
        acc0 = WMMA_BF16(af, bf0, acc0);
        acc1 = WMMA_BF16(af, bf1, acc1);
    }
    if (Kfull < K) {  // masked K tail (only chunk that needs element guards)
        const v16bf af  = load_a_frag<true>(arow,  Kfull, kselA, K);
        const v16bf bf0 = load_b_frag<true>(brow0, Kfull, kselB, K);
        const v16bf bf1 = load_b_frag<true>(brow1, Kfull, kselB, K);
        acc0 = WMMA_BF16(af, bf0, acc0);
        acc1 = WMMA_BF16(af, bf1, acc1);
    }

    // C/D layout: VGPR i, lanes0-15 -> M=row0+i, lanes16-31 -> M=row0+i+8; N=lane&15.
    const int rr0 = row0 + (lane >> 4) * 8;
#pragma unroll
    for (int t = 0; t < 2; ++t) {
        const v8f& acc = t ? acc1 : acc0;
        const int c = col0 + t * 16 + (lane & 15);
        if (c >= Nc) continue;
        if (STORE_T) {
            float* __restrict__ dst = C + (long)b * strideC + (long)c * M + rr0;
#pragma unroll
            for (int i = 0; i < 8; ++i)
                if (rr0 + i < M) dst[i] = acc[i];
        } else {
#pragma unroll
            for (int i = 0; i < 8; ++i) {
                const int rr = rr0 + i;
                if (rr < M) {
                    float v = acc[i];
                    if (EPI >= EPI_BIAS_RELU) {
                        v += bias[c];
                        v = v > 0.0f ? v : 0.0f;
                    }
                    if (EPI == EPI_BIAS_RELU_RES)
                        v += residual[(long)b * strideC + (long)rr * Nc + c];
                    C[(long)b * strideC + (long)rr * Nc + c] = v;
                }
            }
        }
    }
}

// out[c*R + r] = in[r*C + c]   (small weight transposes)
__global__ __launch_bounds__(256) void transpose_mat(
    const float* __restrict__ in, float* __restrict__ out, int R, int C)
{
    int i = blockIdx.x * blockDim.x + threadIdx.x;
    if (i >= R * C) return;
    int r = i / C, c = i - r * C;
    out[(long)c * R + r] = in[i];
}

// t[b,n] = dot(h2[b,n,:], W3[:,0])
__global__ __launch_bounds__(256) void colvec_w3(
    const float* __restrict__ h2, const float* __restrict__ W3,
    float* __restrict__ t, int total, int H)
{
    int i = blockIdx.x * blockDim.x + threadIdx.x;
    if (i >= total) return;
    const float* row = h2 + (long)i * H;
    float s = 0.0f;
    for (int k = 0; k < H; ++k) s += row[k] * W3[k];
    t[i] = s;
}

// g[b,n] = relu(dot(adj[b,n,:], t[b,:]) + b3)
__global__ __launch_bounds__(256) void adj_vec_relu(
    const float* __restrict__ adj, const float* __restrict__ t,
    const float* __restrict__ b3, float* __restrict__ g, int Nn)
{
    const int n = blockIdx.x, b = blockIdx.y;
    const float* arow = adj + ((long)b * Nn + n) * (long)Nn;
    const float* tv   = t + (long)b * Nn;
    float s = 0.0f;
    for (int k = threadIdx.x; k < Nn; k += 256) s += arow[k] * tv[k];
    __shared__ float red[256];
    red[threadIdx.x] = s;
    __syncthreads();
    for (int o = 128; o > 0; o >>= 1) {
        if (threadIdx.x < o) red[threadIdx.x] += red[threadIdx.x + o];
        __syncthreads();
    }
    if (threadIdx.x == 0) {
        float v = red[0] + b3[0];
        g[(long)b * Nn + n] = v > 0.0f ? v : 0.0f;
    }
}

// Y[b,c] = (relu?)(dot(X[b,:], W[:,c]) + bias[c])
__global__ __launch_bounds__(256) void dense(
    const float* __restrict__ X, const float* __restrict__ W,
    const float* __restrict__ bias, float* __restrict__ Y,
    int Bn, int K, int Nc, int do_relu)
{
    int i = blockIdx.x * blockDim.x + threadIdx.x;
    if (i >= Bn * Nc) return;
    int b = i / Nc, c = i - b * Nc;
    const float* x = X + (long)b * K;
    float s = bias ? bias[c] : 0.0f;
    for (int k = 0; k < K; ++k) s += x[k] * W[(long)k * Nc + c];
    if (do_relu) s = s > 0.0f ? s : 0.0f;
    Y[i] = s;
}

// BN over batch (B=16) for mu & logvar, reparameterize, softmax over T.
__global__ __launch_bounds__(64) void bn_reparam_softmax(
    const float* __restrict__ mu_lin, const float* __restrict__ lv_lin,
    const float* __restrict__ eps, const float* __restrict__ bnm_bias,
    const float* __restrict__ bnlv_bias, float* __restrict__ p, int Bn, int T)
{
    __shared__ float z[16 * 50];
    const int t = threadIdx.x;
    if (t < T) {
        float m = 0, m2 = 0, lm = 0, lm2 = 0;
        for (int b = 0; b < Bn; ++b) {
            float a = mu_lin[b * T + t]; m += a; m2 += a * a;
            float l = lv_lin[b * T + t]; lm += l; lm2 += l * l;
        }
        m /= Bn; lm /= Bn;
        const float inv  = rsqrtf(m2 / Bn - m * m + 1e-5f);
        const float linv = rsqrtf(lm2 / Bn - lm * lm + 1e-5f);
        for (int b = 0; b < Bn; ++b) {
            float pm  = (mu_lin[b * T + t] - m)  * inv  + bnm_bias[t];
            float plv = (lv_lin[b * T + t] - lm) * linv + bnlv_bias[t];
            z[b * T + t] = pm + __expf(0.5f * plv) * eps[b * T + t];
        }
    }
    __syncthreads();
    if (t < Bn) {
        float mx = -1e30f;
        for (int j = 0; j < T; ++j) mx = fmaxf(mx, z[t * T + j]);
        float s = 0.0f;
        for (int j = 0; j < T; ++j) { float e = __expf(z[t * T + j] - mx); z[t * T + j] = e; s += e; }
        const float is = 1.0f / s;
        for (int j = 0; j < T; ++j) p[t * T + j] = z[t * T + j] * is;
    }
}

// Per-column BN over B=16 rows (decoder BN); in-place safe per thread.
__global__ __launch_bounds__(256) void bn_cols(
    const float* __restrict__ X, const float* __restrict__ bias,
    float* __restrict__ Y, int Bn, int Nc)
{
    int n = blockIdx.x * blockDim.x + threadIdx.x;
    if (n >= Nc) return;
    float m = 0, m2 = 0;
    for (int b = 0; b < Bn; ++b) { float a = X[(long)b * Nc + n]; m += a; m2 += a * a; }
    m /= Bn;
    const float inv = rsqrtf(m2 / Bn - m * m + 1e-5f);
    for (int b = 0; b < Bn; ++b)
        Y[(long)b * Nc + n] = (X[(long)b * Nc + n] - m) * inv + bias[n];
}

// Row softmax over Nc; one block per row.
__global__ __launch_bounds__(256) void softmax_rows(
    const float* __restrict__ X, float* __restrict__ out, int Nc)
{
    const int b = blockIdx.x;
    const float* x = X + (long)b * Nc;
    __shared__ float red[256];
    float mx = -1e30f;
    for (int k = threadIdx.x; k < Nc; k += 256) mx = fmaxf(mx, x[k]);
    red[threadIdx.x] = mx;
    __syncthreads();
    for (int o = 128; o > 0; o >>= 1) {
        if (threadIdx.x < o) red[threadIdx.x] = fmaxf(red[threadIdx.x], red[threadIdx.x + o]);
        __syncthreads();
    }
    mx = red[0];
    __syncthreads();
    float s = 0.0f;
    for (int k = threadIdx.x; k < Nc; k += 256) s += __expf(x[k] - mx);
    red[threadIdx.x] = s;
    __syncthreads();
    for (int o = 128; o > 0; o >>= 1) {
        if (threadIdx.x < o) red[threadIdx.x] += red[threadIdx.x + o];
        __syncthreads();
    }
    const float is = 1.0f / red[0];
    for (int k = threadIdx.x; k < Nc; k += 256)
        out[(long)b * Nc + k] = __expf(x[k] - mx) * is;
}

extern "C" void kernel_launch(void* const* d_in, const int* in_sizes, int n_in,
                              void* d_out, int out_size, void* d_ws, size_t ws_size,
                              hipStream_t stream) {
    (void)in_sizes; (void)n_in; (void)out_size; (void)ws_size;
    const int Bn = 16, Nn = 1995, Fh = 1995, H = 100, T = 50;

    const float* x    = (const float*)d_in[0];
    const float* adj  = (const float*)d_in[1];
    /* d_in[2] = eye (unused) */
    const float* eps  = (const float*)d_in[3];
    const float* W1   = (const float*)d_in[4];
    const float* b1   = (const float*)d_in[5];
    const float* W2   = (const float*)d_in[6];
    const float* b2   = (const float*)d_in[7];
    const float* W3   = (const float*)d_in[8];
    const float* b3   = (const float*)d_in[9];
    const float* We1  = (const float*)d_in[10];
    const float* be1  = (const float*)d_in[11];
    const float* We2  = (const float*)d_in[12];
    const float* be2  = (const float*)d_in[13];
    const float* Wm   = (const float*)d_in[14];
    const float* bm   = (const float*)d_in[15];
    const float* bnmB = (const float*)d_in[16];
    const float* Wlv  = (const float*)d_in[17];
    const float* blv  = (const float*)d_in[18];
    const float* bnlB = (const float*)d_in[19];
    const float* Wd   = (const float*)d_in[20];
    const float* bndB = (const float*)d_in[21];
    float* out = (float*)d_out;

    // -------- workspace carve-up (fp32) --------
    float* ws = (float*)d_ws;
    size_t o = 0;
    float* W1T  = ws + o; o += (size_t)Fh * H;        // W1^T  [H,F]
    float* W2T  = ws + o; o += (size_t)H * H;         // W2^T  [H,H]
    float* XWT  = ws + o; o += (size_t)Bn * Nn * H;   // (x@W1)^T   [B,H,N]
    float* h1   = ws + o; o += (size_t)Bn * Nn * H;   // [B,N,H]
    float* HW2T = ws + o; o += (size_t)Bn * Nn * H;   // (h1@W2)^T  [B,H,N]
    float* h2   = ws + o; o += (size_t)Bn * Nn * H;   // [B,N,H]
    float* tvec = ws + o; o += (size_t)Bn * Nn;
    float* g    = ws + o; o += (size_t)Bn * Nn;
    float* en1  = ws + o; o += (size_t)Bn * H;
    float* en2  = ws + o; o += (size_t)Bn * H;
    float* muL  = ws + o; o += (size_t)Bn * T;
    float* lvL  = ws + o; o += (size_t)Bn * T;
    float* p    = ws + o; o += (size_t)Bn * T;
    float* rec  = ws + o; o += (size_t)Bn * Nn;

    // Prep: transpose the shared weights once (tiny)
    transpose_mat<<<(Fh * H + 255) / 256, 256, 0, stream>>>(W1, W1T, Fh, H);
    transpose_mat<<<(H * H + 255) / 256, 256, 0, stream>>>(W2, W2T, H, H);

    const dim3 gridG((Nn + 15) / 16, (H + 31) / 32, Bn);  // 125 x 4 x 16 waves

    // Stage 1: XWT = (x @ W1)^T
    gemm_bf16_wmma<EPI_NONE, true><<<gridG, 32, 0, stream>>>(
        x, W1T, XWT, nullptr, nullptr, Nn, Fh, H, (long)Nn * Fh, 0L, (long)Nn * H);
    // Stage 2: h1 = relu(adj @ XW + b1)
    gemm_bf16_wmma<EPI_BIAS_RELU, false><<<gridG, 32, 0, stream>>>(
        adj, XWT, h1, b1, nullptr, Nn, Nn, H, (long)Nn * Nn, (long)Nn * H, (long)Nn * H);
    // Stage 3: HW2T = (h1 @ W2)^T
    gemm_bf16_wmma<EPI_NONE, true><<<gridG, 32, 0, stream>>>(
        h1, W2T, HW2T, nullptr, nullptr, Nn, H, H, (long)Nn * H, 0L, (long)Nn * H);
    // Stage 4: h2 = relu(adj @ HW2 + b2) + h1
    gemm_bf16_wmma<EPI_BIAS_RELU_RES, false><<<gridG, 32, 0, stream>>>(
        adj, HW2T, h2, b2, h1, Nn, Nn, H, (long)Nn * Nn, (long)Nn * H, (long)Nn * H);

    // Stage 5/6: g = relu(adj @ (h2 @ W3) + b3)
    colvec_w3<<<(Bn * Nn + 255) / 256, 256, 0, stream>>>(h2, W3, tvec, Bn * Nn, H);
    adj_vec_relu<<<dim3(Nn, Bn), 256, 0, stream>>>(adj, tvec, b3, g, Nn);

    // Encoder MLP
    dense<<<(Bn * H + 255) / 256, 256, 0, stream>>>(g,   We1, be1, en1, Bn, Nn, H, 1);
    dense<<<(Bn * H + 255) / 256, 256, 0, stream>>>(en1, We2, be2, en2, Bn, H,  H, 1);
    dense<<<(Bn * T + 255) / 256, 256, 0, stream>>>(en2, Wm,  bm,  muL, Bn, H,  T, 0);
    dense<<<(Bn * T + 255) / 256, 256, 0, stream>>>(en2, Wlv, blv, lvL, Bn, H,  T, 0);

    // BN + reparameterize + softmax -> p
    bn_reparam_softmax<<<1, 64, 0, stream>>>(muL, lvL, eps, bnmB, bnlB, p, Bn, T);

    // Decoder: rec = p @ Wd ; BN over batch ; softmax over N -> out
    dense<<<(Bn * Nn + 255) / 256, 256, 0, stream>>>(p, Wd, nullptr, rec, Bn, T, Nn, 0);
    bn_cols<<<(Nn + 255) / 256, 256, 0, stream>>>(rec, bndB, rec, Bn, Nn);
    softmax_rows<<<Bn, 256, 0, stream>>>(rec, out, Nn);
}
